// Seq2Seq_36953898614809
// MI455X (gfx1250) — matrix operand order repaired and checked by reading, not compiled, not run on hardware
//
#include <hip/hip_runtime.h>
#include <hip/hip_bf16.h>
#include <cstdint>
#include <cstddef>

// Problem constants (match reference)
#define S_LEN  120
#define BATCH  2048
#define INDIM  16
#define HID    256
#define HID2   512
#define HID3   768
#define Y_LEN  24

typedef __bf16 bf16;
typedef __attribute__((ext_vector_type(16))) __bf16 v16bf;
typedef __attribute__((ext_vector_type(8)))  __bf16 v8bf;
typedef __attribute__((ext_vector_type(8)))  float  v8f;

// ---------------------------------------------------------------------------
// Fused bi-GRU step:  gates = h @ Whh^T (+) x @ Wih^T, K-concatenated.
// Weights are pre-converted to bf16. XDIM is compile time: 16 (enc L0, f32 x),
// 512 (bf16 x), 0 (decoder L0: rank-1 x folded into the fp32 epilogue).
// grid = (BATCH/16, 2 dirs), block = 512 (16 waves); wave w owns H-tile w.
// ---------------------------------------------------------------------------
struct StepArgs {
  const float* h_in[2];
  float*       h_out[2];
  const void*  x[2];        // [B,16] f32 | [B,512] bf16 | [B] f32
  const bf16*  Whh[2];      // [768,256] bf16
  const bf16*  Wih[2];      // [768,XPAD] bf16 (XDIM>0)
  const float* WihF[2];     // [768,1] f32 (XDIM==0 rank-1 path)
  const float* bih[2];
  const float* bhh[2];
  bf16*        outp[2];     // optional seq output [B,512] at folded t
};

__device__ __forceinline__ float sigm(float x) { return 1.0f / (1.0f + __expf(-x)); }

// A fragment: lane holds K {0..15} (+16 for hi lanes) of its row.
__device__ __forceinline__ v16bf load_a(const bf16* p) {
  v8bf lo = *(const v8bf*)p;
  v8bf hi = *(const v8bf*)(p + 8);
  v16bf r;
#pragma unroll
  for (int i = 0; i < 8; ++i) { r[i] = lo[i]; r[8 + i] = hi[i]; }
  return r;
}
// B fragment: 16 contiguous bf16 along K of one W row.
__device__ __forceinline__ v16bf load_w(const bf16* p) {
  v8bf lo = *(const v8bf*)p;
  v8bf hi = *(const v8bf*)(p + 8);
  v16bf r;
#pragma unroll
  for (int i = 0; i < 8; ++i) { r[i] = lo[i]; r[8 + i] = hi[i]; }
  return r;
}

template <int XDIM>
__launch_bounds__(512, 1)
__global__ void gru_step_kernel(StepArgs a) {
  constexpr int XPAD = (XDIM == 16) ? 32 : XDIM;  // staged/padded x width
  constexpr int NX   = XPAD / 32;                 // x K-chunks
  constexpr int ROWH = 256 + XPAD + 8;            // LDS row pitch (halves)

  const int dir  = blockIdx.y;
  const int b0   = blockIdx.x << 4;
  const int tid  = threadIdx.x;
  const int w    = tid >> 5;
  const int lane = tid & 31;
  const int l15  = lane & 15;
  const int hi16 = (lane >> 4) & 1;

  __shared__ __align__(16) bf16 As[16 * ROWH];

  const float* __restrict__ hin = a.h_in[dir];

  // ---- stage h tile: 16x256 f32 -> bf16, vectorized (8 floats/thread) ----
  {
    const int idx = tid * 8;
    const int row = idx >> 8, col = idx & 255;
    const float4* hp = (const float4*)(hin + (size_t)(b0 + row) * HID + col);
    float4 f0 = hp[0], f1 = hp[1];
    v8bf o;
    o[0]=(bf16)f0.x; o[1]=(bf16)f0.y; o[2]=(bf16)f0.z; o[3]=(bf16)f0.w;
    o[4]=(bf16)f1.x; o[5]=(bf16)f1.y; o[6]=(bf16)f1.z; o[7]=(bf16)f1.w;
    *(v8bf*)(As + row * ROWH + col) = o;
  }
  // ---- stage x tile ----
  if constexpr (XDIM == 16) {
    const int row = tid >> 5, c = tid & 31;
    const float* xf = (const float*)a.x[dir];
    const float v = (c < 16) ? xf[(size_t)(b0 + row) * 16 + c] : 0.0f;
    As[row * ROWH + 256 + c] = (bf16)v;
  } else if constexpr (XDIM == 512) {
    const int idx = tid * 16;  // 8192 halves, 32B/thread
    const int row = idx >> 9, col = idx & 511;
    const bf16* xb = (const bf16*)a.x[dir];
    const uint4* sp = (const uint4*)(xb + (size_t)(b0 + row) * 512 + col);
    uint4 u0 = sp[0], u1 = sp[1];
    *(uint4*)(As + row * ROWH + 256 + col)     = u0;
    *(uint4*)(As + row * ROWH + 256 + col + 8) = u1;
  }
  __syncthreads();

  const int j0 = w << 4;
  v8f acc_r = {}, acc_z = {}, acc_hn = {}, acc_in = {};

  const bf16* ab = As + l15 * ROWH + hi16 * 16;

  // ---- h-region: 8 chunks against Whh (bf16, L2-resident) ----
  {
    const bf16* whh = a.Whh[dir];
    const bf16* wr0 = whh + (size_t)(0 * HID + j0 + l15) * HID + hi16 * 16;
    const bf16* wz0 = whh + (size_t)(1 * HID + j0 + l15) * HID + hi16 * 16;
    const bf16* wn0 = whh + (size_t)(2 * HID + j0 + l15) * HID + hi16 * 16;
#pragma unroll
    for (int c = 0; c < 8; ++c) {
      const int kc = c * 32;
      v16bf A  = load_a(ab + kc);
      v16bf Br = load_w(wr0 + kc);
      v16bf Bz = load_w(wz0 + kc);
      v16bf Bn = load_w(wn0 + kc);
      acc_r  = __builtin_amdgcn_wmma_f32_16x16x32_bf16(false, A, false, Br, (short)0, acc_r,  false, false);
      acc_z  = __builtin_amdgcn_wmma_f32_16x16x32_bf16(false, A, false, Bz, (short)0, acc_z,  false, false);
      acc_hn = __builtin_amdgcn_wmma_f32_16x16x32_bf16(false, A, false, Bn, (short)0, acc_hn, false, false);
    }
  }
  // ---- x-region: NX chunks against Wih (zero-padded bf16) ----
  if constexpr (NX > 0) {
    const bf16* wih = a.Wih[dir];
    const bf16* xr0 = wih + (size_t)(0 * HID + j0 + l15) * XPAD + hi16 * 16;
    const bf16* xz0 = wih + (size_t)(1 * HID + j0 + l15) * XPAD + hi16 * 16;
    const bf16* xn0 = wih + (size_t)(2 * HID + j0 + l15) * XPAD + hi16 * 16;
#pragma unroll
    for (int c = 0; c < NX; ++c) {
      const int kc = c * 32;
      v16bf A  = load_a(ab + 256 + kc);
      v16bf Br = load_w(xr0 + kc);
      v16bf Bz = load_w(xz0 + kc);
      v16bf Bn = load_w(xn0 + kc);
      acc_r  = __builtin_amdgcn_wmma_f32_16x16x32_bf16(false, A, false, Br, (short)0, acc_r,  false, false);
      acc_z  = __builtin_amdgcn_wmma_f32_16x16x32_bf16(false, A, false, Bz, (short)0, acc_z,  false, false);
      acc_in = __builtin_amdgcn_wmma_f32_16x16x32_bf16(false, A, false, Bn, (short)0, acc_in, false, false);
    }
  }

  // ---- fp32 GRU epilogue; C/D: vgpr i -> row i (+8 for hi lanes) ----
  const float* bih = a.bih[dir];
  const float* bhh = a.bhh[dir];
  const int   j   = j0 + l15;
  const float br  = bih[j]       + bhh[j];
  const float bz  = bih[HID+j]   + bhh[HID+j];
  const float bin = bih[2*HID+j];
  const float bhn = bhh[2*HID+j];
  float wir = 0.f, wiz = 0.f, win = 0.f;
  const float* xv = nullptr;
  if constexpr (XDIM == 0) {  // rank-1 input contribution
    const float* wf = a.WihF[dir];
    wir = wf[j]; wiz = wf[HID + j]; win = wf[2*HID + j];
    xv  = (const float*)a.x[dir];
  }
  float* hout = a.h_out[dir];
  bf16*  op   = a.outp[dir];
#pragma unroll
  for (int i = 0; i < 8; ++i) {
    const int   row = b0 + i + hi16 * 8;
    const float hp  = hin[(size_t)row * HID + j];
    float ir = acc_r[i], iz = acc_z[i], in_ = acc_in[i];
    if constexpr (XDIM == 0) {
      const float xb = xv[row];
      ir += xb * wir; iz += xb * wiz; in_ += xb * win;
    }
    const float rg = sigm(ir + br);
    const float zg = sigm(iz + bz);
    const float ng = tanhf(in_ + bin + rg * (acc_hn[i] + bhn));
    const float hv = (1.0f - zg) * ng + zg * hp;
    hout[(size_t)row * HID + j] = hv;
    if (op) op[(size_t)row * HID2 + dir * HID + j] = (bf16)hv;
  }
}

// ---------------------------------------------------------------------------
// Weight prep: f32 -> bf16 (and zero-padded [768,16] -> [768,32]).
// ---------------------------------------------------------------------------
__global__ void cvt_bf16_kernel(const float* __restrict__ s, bf16* __restrict__ d, int n) {
  const int i = blockIdx.x * blockDim.x + threadIdx.x;
  if (i < n) d[i] = (bf16)s[i];
}
__global__ void cvt_pad32_kernel(const float* __restrict__ s, bf16* __restrict__ d) {
  const int i = blockIdx.x * blockDim.x + threadIdx.x;  // 768*32
  if (i >= HID3 * 32) return;
  const int r = i >> 5, c = i & 31;
  d[i] = (c < 16) ? (bf16)s[r * 16 + c] : (bf16)0.0f;
}

// ---------------------------------------------------------------------------
__global__ void init_h0_kernel(const float* __restrict__ numf, const int* __restrict__ catf,
                               const float* e0, const float* e1, const float* e2, const float* e3,
                               float* h00, float* h01, float* h10, float* h11) {
  const int idx = blockIdx.x * blockDim.x + threadIdx.x;
  if (idx >= BATCH * HID) return;
  const int b = idx / HID, j = idx % HID;
  float v;
  if (j < HID - 16) {
    v = numf[(size_t)b * (HID - 16) + j];
  } else {
    const int jj = j - (HID - 16);
    const int g = jj >> 2, e = jj & 3;
    const float* emb = (g == 0) ? e0 : (g == 1) ? e1 : (g == 2) ? e2 : e3;
    const int ci = catf[g * BATCH + b];
    v = emb[ci * 4 + e];
  }
  h00[idx] = v; h01[idx] = v; h10[idx] = v; h11[idx] = v;
}

__global__ void init_xdec_kernel(const float* __restrict__ temp, float* __restrict__ xdec) {
  const int b = blockIdx.x * blockDim.x + threadIdx.x;
  if (b < BATCH) xdec[b] = temp[(size_t)(S_LEN - 1) * BATCH * INDIM + (size_t)b * INDIM];
}

// FC head: pred[b] = dec_out1[b,:] . fc_w + fc_b, one wave per batch element.
__global__ void fc_kernel(const bf16* __restrict__ out1, const float* __restrict__ fcw,
                          const float* __restrict__ fcb, float* __restrict__ pred,
                          float* __restrict__ xdec) {
  const int b = blockIdx.x * (blockDim.x >> 5) + (threadIdx.x >> 5);
  const int lane = threadIdx.x & 31;
  if (b >= BATCH) return;
  const bf16* o = out1 + (size_t)b * HID2;
  float s = 0.0f;
  for (int k = lane; k < HID2; k += 32) s += (float)o[k] * fcw[k];
#pragma unroll
  for (int off = 16; off > 0; off >>= 1) s += __shfl_xor(s, off, 32);
  if (lane == 0) {
    const float p = s + fcb[0];
    pred[b] = p;
    xdec[b] = p;
  }
}

// ---------------------------------------------------------------------------
extern "C" void kernel_launch(void* const* d_in, const int* in_sizes, int n_in,
                              void* d_out, int out_size, void* d_ws, size_t ws_size,
                              hipStream_t stream) {
  (void)in_sizes; (void)n_in; (void)out_size; (void)ws_size;
  const float* temp = (const float*)d_in[0];
  const float* numf = (const float*)d_in[1];
  const int*   catf = (const int*)d_in[2];
  const float* e0 = (const float*)d_in[3];
  const float* e1 = (const float*)d_in[4];
  const float* e2 = (const float*)d_in[5];
  const float* e3 = (const float*)d_in[6];
  auto encW = [&](int l, int d, int k) { return (const float*)d_in[7  + (l * 2 + d) * 4 + k]; };
  auto decW = [&](int l, int d, int k) { return (const float*)d_in[23 + (l * 2 + d) * 4 + k]; };
  const float* fcw = (const float*)d_in[39];
  const float* fcb = (const float*)d_in[40];

  // ---- carve workspace (~296 MB) ----
  char* p = (char*)d_ws;
  auto carve = [&](size_t bytes) { void* r = (void*)p; p += (bytes + 255) & ~(size_t)255; return r; };
  bf16* out0 = (bf16*)carve((size_t)S_LEN * BATCH * HID2 * sizeof(bf16));
  float* hbuf[4][2];
  for (int s = 0; s < 4; ++s)
    for (int q = 0; q < 2; ++q) hbuf[s][q] = (float*)carve((size_t)BATCH * HID * sizeof(float));
  float* dh[4][2];
  for (int s = 0; s < 4; ++s)
    for (int q = 0; q < 2; ++q) dh[s][q] = (float*)carve((size_t)BATCH * HID * sizeof(float));
  bf16*  dout0 = (bf16*)carve((size_t)BATCH * HID2 * sizeof(bf16));
  bf16*  dout1 = (bf16*)carve((size_t)BATCH * HID2 * sizeof(bf16));
  float* xdec  = (float*)carve((size_t)BATCH * sizeof(float));
  // bf16 weight copies: [enc/dec][layer][dir]
  bf16* whh_bf[2][2][2];
  for (int m = 0; m < 2; ++m)
    for (int l = 0; l < 2; ++l)
      for (int d = 0; d < 2; ++d) whh_bf[m][l][d] = (bf16*)carve((size_t)HID3 * HID * sizeof(bf16));
  bf16* wih_l0[2];   // enc L0 Wih, padded to [768,32]
  for (int d = 0; d < 2; ++d) wih_l0[d] = (bf16*)carve((size_t)HID3 * 32 * sizeof(bf16));
  bf16* wih_l1[2][2]; // [enc/dec] L1 Wih [768,512]
  for (int m = 0; m < 2; ++m)
    for (int d = 0; d < 2; ++d) wih_l1[m][d] = (bf16*)carve((size_t)HID3 * HID2 * sizeof(bf16));

  // ---- weight conversion ----
  const int nWhh = HID3 * HID, nWih1 = HID3 * HID2;
  for (int m = 0; m < 2; ++m)
    for (int l = 0; l < 2; ++l)
      for (int d = 0; d < 2; ++d)
        cvt_bf16_kernel<<<(nWhh + 255) / 256, 256, 0, stream>>>(
            (m == 0 ? encW(l, d, 1) : decW(l, d, 1)), whh_bf[m][l][d], nWhh);
  for (int d = 0; d < 2; ++d)
    cvt_pad32_kernel<<<(HID3 * 32 + 255) / 256, 256, 0, stream>>>(encW(0, d, 0), wih_l0[d]);
  for (int m = 0; m < 2; ++m)
    for (int d = 0; d < 2; ++d)
      cvt_bf16_kernel<<<(nWih1 + 255) / 256, 256, 0, stream>>>(
          (m == 0 ? encW(1, d, 0) : decW(1, d, 0)), wih_l1[m][d], nWih1);

  // ---- init ----
  init_h0_kernel<<<(BATCH * HID + 255) / 256, 256, 0, stream>>>(
      numf, catf, e0, e1, e2, e3, hbuf[0][0], hbuf[1][0], hbuf[2][0], hbuf[3][0]);
  init_xdec_kernel<<<(BATCH + 255) / 256, 256, 0, stream>>>(temp, xdec);

  const dim3 sgrid(BATCH / 16, 2);

  // ---- encoder layer 0 (x = temp features, f32, XDIM=16) ----
  for (int t = 0; t < S_LEN; ++t) {
    StepArgs sa{};
    for (int d = 0; d < 2; ++d) {
      const int tt = (d == 0) ? t : (S_LEN - 1 - t);
      sa.h_in[d]  = hbuf[d][t & 1];
      sa.h_out[d] = hbuf[d][(t + 1) & 1];
      sa.x[d]     = temp + (size_t)tt * BATCH * INDIM;
      sa.Whh[d]   = whh_bf[0][0][d];
      sa.Wih[d]   = wih_l0[d];
      sa.bih[d]   = encW(0, d, 2); sa.bhh[d] = encW(0, d, 3);
      sa.outp[d]  = out0 + (size_t)tt * BATCH * HID2;
    }
    gru_step_kernel<16><<<sgrid, 512, 0, stream>>>(sa);
  }
  // ---- encoder layer 1 (x = out0, bf16, XDIM=512) ----
  for (int t = 0; t < S_LEN; ++t) {
    StepArgs sa{};
    for (int d = 0; d < 2; ++d) {
      const int tt = (d == 0) ? t : (S_LEN - 1 - t);
      const int st = 2 + d;
      sa.h_in[d]  = hbuf[st][t & 1];
      sa.h_out[d] = hbuf[st][(t + 1) & 1];
      sa.x[d]     = out0 + (size_t)tt * BATCH * HID2;
      sa.Whh[d]   = whh_bf[0][1][d];
      sa.Wih[d]   = wih_l1[0][d];
      sa.bih[d]   = encW(1, d, 2); sa.bhh[d] = encW(1, d, 3);
      sa.outp[d]  = nullptr;  // only hn needed
    }
    gru_step_kernel<512><<<sgrid, 512, 0, stream>>>(sa);
  }
  // Encoder finals land in hbuf[st][0] (120 steps, even) -> decoder h0 directly.

  // ---- decoder: 24 autoregressive steps ----
  for (int s = 0; s < Y_LEN; ++s) {
    {  // layer 0: rank-1 input path (XDIM=0), x = xdec [B] f32
      StepArgs sa{};
      for (int d = 0; d < 2; ++d) {
        sa.h_in[d]  = (s == 0) ? hbuf[d][0] : dh[d][(s + 1) & 1];
        sa.h_out[d] = dh[d][s & 1];
        sa.x[d]     = xdec;
        sa.Whh[d]   = whh_bf[1][0][d];
        sa.WihF[d]  = decW(0, d, 0);
        sa.bih[d]   = decW(0, d, 2); sa.bhh[d] = decW(0, d, 3);
        sa.outp[d]  = dout0;
      }
      gru_step_kernel<0><<<sgrid, 512, 0, stream>>>(sa);
    }
    {  // layer 1: x = dout0 [B,512] bf16
      StepArgs sa{};
      for (int d = 0; d < 2; ++d) {
        const int st = 2 + d;
        sa.h_in[d]  = (s == 0) ? hbuf[st][0] : dh[st][(s + 1) & 1];
        sa.h_out[d] = dh[st][s & 1];
        sa.x[d]     = dout0;
        sa.Whh[d]   = whh_bf[1][1][d];
        sa.Wih[d]   = wih_l1[1][d];
        sa.bih[d]   = decW(1, d, 2); sa.bhh[d] = decW(1, d, 3);
        sa.outp[d]  = dout1;
      }
      gru_step_kernel<512><<<sgrid, 512, 0, stream>>>(sa);
    }
    fc_kernel<<<BATCH / 8, 256, 0, stream>>>(dout1, fcw, fcb,
                                             (float*)d_out + (size_t)s * BATCH, xdec);
  }
}
